// AntiPatternLoss_9483287789850
// MI455X (gfx1250) — 8-bit matrix operand order repaired and checked by reading, compiled, not run on hardware
//
#include <hip/hip_runtime.h>
#include <hip/hip_bf16.h>
#include <stdint.h>


typedef int v8i __attribute__((ext_vector_type(8)));

#define B_CONST   16
#define T_CONST   2048
#define V_CONST   128
#define NGRAM_C   3
#define L_CONST   (T_CONST - NGRAM_C + 1)
#define NT_CONST  ((L_CONST + 15) / 16)
#define TJ_PER_WAVE 4
#define CPAD      2048
#define REP_PEN_C 1.2f
#define MATCH_SUM 9

__global__ void __launch_bounds__(256) argmax_softmax_kernel(
    const float* __restrict__ logits, int* __restrict__ pred,
    float* __restrict__ ppred)
{
    const int row  = blockIdx.x * 8 + (threadIdx.x >> 5);
    const int lane = threadIdx.x & 31;
    if (row >= B_CONST * T_CONST) return;

    const float4* rp = reinterpret_cast<const float4*>(
        logits + (size_t)row * V_CONST);
    float4 v = rp[lane];

    float m  = v.x; int mi = lane * 4;
    if (v.y > m) { m = v.y; mi = lane * 4 + 1; }
    if (v.z > m) { m = v.z; mi = lane * 4 + 2; }
    if (v.w > m) { m = v.w; mi = lane * 4 + 3; }

#pragma unroll
    for (int off = 16; off > 0; off >>= 1) {
        float om  = __shfl_xor(m, off, 32);
        int   omi = __shfl_xor(mi, off, 32);
        if (om > m || (om == m && omi < mi)) { m = om; mi = omi; }
    }

    float s = __expf(v.x - m) + __expf(v.y - m) +
              __expf(v.z - m) + __expf(v.w - m);
#pragma unroll
    for (int off = 16; off > 0; off >>= 1) s += __shfl_xor(s, off, 32);

    if (lane == 0) {
        pred[row]  = mi;
        ppred[row] = 1.0f / s;
    }
}

__device__ __forceinline__ void build_words(
    const int* __restrict__ predp, int idx, bool valid, unsigned W[3][4])
{
#pragma unroll
    for (int t = 0; t < NGRAM_C; ++t) {
        const int tok = valid ? predp[idx + t] : 0;
        const unsigned w0 = 1u << ((tok & 3) * 8);
        const unsigned w1 = 1u << (((tok >> 2) & 3) * 8);
        const int      f2 = (tok >> 4) & 7;
        const unsigned w2 = 1u << ((f2 & 3) * 8);
        W[t][0] = valid ? w0 : 0u;
        W[t][1] = valid ? w1 : 0u;
        W[t][2] = (valid && f2 <  4) ? w2 : 0u;
        W[t][3] = (valid && f2 >= 4) ? w2 : 0u;
    }
}

__global__ void __launch_bounds__(32) trigram_wmma_kernel(
    const int* __restrict__ pred, unsigned* __restrict__ counts)
{
    const int ti   = blockIdx.y;
    const int tj0  = blockIdx.x * TJ_PER_WAVE;
    if (tj0 + TJ_PER_WAVE - 1 < ti) return;
    const int b    = blockIdx.z;
    const int lane = threadIdx.x;
    const int half = lane >> 4;
    const int lm   = lane & 15;

    const int* predb = pred + b * T_CONST;

    const int ia = ti * 16 + lm;
    unsigned WA[3][4];
    build_words(predb, ia, ia < L_CONST, WA);

    v8i A;
    A[0] = (int)(half ? WA[0][2] : WA[0][0]);
    A[1] = (int)(half ? WA[0][3] : WA[0][1]);
    A[2] = (int)(half ? WA[1][2] : WA[1][0]);
    A[3] = (int)(half ? WA[1][3] : WA[1][1]);
    A[4] = (int)(half ? WA[2][2] : WA[2][0]);
    A[5] = (int)(half ? WA[2][3] : WA[2][1]);
    A[6] = 0;
    A[7] = 0;

#pragma unroll
    for (int s = 0; s < TJ_PER_WAVE; ++s) {
        const int tj = tj0 + s;
        if (tj < ti) continue;

        const int jb = tj * 16 + lm;
        unsigned WB[3][4];
        build_words(predb, jb, jb < L_CONST, WB);

        v8i Bm;
        Bm[0] = (int)(half ? WB[0][2] : WB[0][0]);
        Bm[1] = (int)(half ? WB[0][3] : WB[0][1]);
        Bm[2] = (int)(half ? WB[1][2] : WB[1][0]);
        Bm[3] = (int)(half ? WB[1][3] : WB[1][1]);
        Bm[4] = (int)(half ? WB[2][2] : WB[2][0]);
        Bm[5] = (int)(half ? WB[2][3] : WB[2][1]);
        Bm[6] = 0;
        Bm[7] = 0;

        v8i acc = {};
        acc = __builtin_amdgcn_wmma_i32_16x16x64_iu8(
            false, A, false, Bm, acc, false, false);

        unsigned cnt = 0;
        const int j = tj * 16 + lm;
#pragma unroll
        for (int r = 0; r < 8; ++r) {
            const int i = ti * 16 + half * 8 + r;
            if (acc[r] == MATCH_SUM && (j - i) >= NGRAM_C) cnt++;
        }
        if (cnt) atomicAdd(&counts[b * CPAD + j], cnt);
    }
}

__global__ void zero_counts_kernel(unsigned* __restrict__ counts) {
    const int idx = blockIdx.x * blockDim.x + threadIdx.x;
    if (idx < B_CONST * CPAD) counts[idx] = 0u;
}

__global__ void __launch_bounds__(256) finalize_kernel(
    const unsigned* __restrict__ counts, const float* __restrict__ ppred,
    float* __restrict__ out)
{
    __shared__ float sm[256];
    float acc = 0.0f;
    for (int idx = threadIdx.x; idx < B_CONST * L_CONST; idx += 256) {
        const int b  = idx / L_CONST;
        const int jj = idx - b * L_CONST;
        const unsigned c = counts[b * CPAD + jj];
        if (c) {
            const float* pp = ppred + b * T_CONST + jj;
            const float pen = (pp[0] + pp[1] + pp[2]) * (1.0f / 3.0f);
            acc += (float)c * pen;
        }
    }
    sm[threadIdx.x] = acc;
    __syncthreads();
    for (int s = 128; s > 0; s >>= 1) {
        if (threadIdx.x < s) sm[threadIdx.x] += sm[threadIdx.x + s];
        __syncthreads();
    }
    if (threadIdx.x == 0)
        out[0] = REP_PEN_C * sm[0] / (float)(B_CONST * T_CONST);
}

extern "C" void kernel_launch(void* const* d_in, const int* in_sizes, int n_in,
                              void* d_out, int out_size, void* d_ws, size_t ws_size,
                              hipStream_t stream) {
    const float* logits = (const float*)d_in[0];
    (void)in_sizes; (void)n_in; (void)out_size; (void)ws_size;

    char*     ws     = (char*)d_ws;
    int*      pred   = (int*)(ws);
    float*    ppred  = (float*)(ws + (size_t)B_CONST * T_CONST * 4);
    unsigned* counts = (unsigned*)(ws + (size_t)B_CONST * T_CONST * 8);

    zero_counts_kernel<<<(B_CONST * CPAD + 255) / 256, 256, 0, stream>>>(counts);

    argmax_softmax_kernel<<<(B_CONST * T_CONST) / 8, 256, 0, stream>>>(
        logits, pred, ppred);

    dim3 grid(NT_CONST / TJ_PER_WAVE, NT_CONST, B_CONST);
    trigram_wmma_kernel<<<grid, 32, 0, stream>>>(pred, counts);

    finalize_kernel<<<1, 256, 0, stream>>>(counts, ppred, (float*)d_out);
}
